// MPNEncoder_58394375356596
// MI455X (gfx1250) — matrix operand order repaired and checked by reading, not compiled, not run on hardware
//
#include <hip/hip_runtime.h>
#include <hip/hip_bf16.h>

// ---------------- problem dims ----------------
#define HH        256
#define SS        4
#define BB        512
#define N_MOLS    2048
#define APM       32
#define N_ATOMS   65536
#define N_BONDS   131072
#define NB1       131073        // N_BONDS + 1 (pad row)
#define MB_PAD    131088        // NB1 padded to multiple of 16 (16*8193)
#define MAX_NB    6
#define AFD       133
#define BFD       147
#define KB_PAD    160           // BOND_FDIM padded to 32
#define KO_PAD    416           // (ATOM_FDIM + H)=389 padded to 32
#define DEPTH     3
#define N_ITERS   3

typedef __attribute__((ext_vector_type(16))) _Float16 v16h;
typedef __attribute__((ext_vector_type(8)))  _Float16 v8h;
typedef __attribute__((ext_vector_type(8)))  float    v8f;

__device__ __forceinline__ float sigm(float x) { return 1.0f / (1.0f + expf(-x)); }

// ---------------------------------------------------------------------------
// f16 WMMA GEMM: C[M x N] = act( A[M x K](lda) @ B[K x N] + bias + resid )
// B staged transposed: Bt[n*K + k] = B[k][n]  (per-lane loads contiguous).
// Each wave owns ONE 16-row M-tile and FOUR 16-col N-tiles: a single A
// fragment feeds 4 WMMAs per k-step (4x less A traffic, 4:5 wmma:load ratio).
// M multiple of 16, K multiple of 32, Ntiles multiple of 4.
// A-frag (16-bit 16x32): lane m=lane&15, kh=lane>>4,
//   elem j -> K=k0+16*kh+j
// B-frag (32x16): lane n=lane&15, kh=lane>>4, elem j -> K=k0+16*kh+j
// C/D: VGPR r -> M = r + 8*kh, N = lane&15
// ---------------------------------------------------------------------------
__global__ __launch_bounds__(256)
void gemm_wmma(const _Float16* __restrict__ A, int lda,
               const _Float16* __restrict__ Bt,
               const float* __restrict__ bias,
               const _Float16* __restrict__ resid,
               float* __restrict__ Cf,
               _Float16* __restrict__ Ch,
               _Float16* __restrict__ Cpre,
               int ldc, int Mtiles, int Ntiles, int K, int do_relu)
{
    int ng   = Ntiles >> 2;                       // N-tile groups of 4
    int wave = (blockIdx.x * blockDim.x + threadIdx.x) >> 5;
    if (wave >= Mtiles * ng) return;
    int mt  = wave / ng;
    int ntg = wave - mt * ng;                     // base N tile = ntg*4
    int lane = threadIdx.x & 31;
    int l15  = lane & 15;
    int kh   = lane >> 4;

    const _Float16* arow = A  + (size_t)(mt * 16 + l15) * lda + (kh << 4);
    const _Float16* brow = Bt + (size_t)(ntg * 64 + l15) * K  + (kh << 4);
    const size_t bstride = (size_t)16 * K;        // halves between N-tiles

    v8f acc0 = {}, acc1 = {}, acc2 = {}, acc3 = {};
    for (int k0 = 0; k0 < K; k0 += 32) {
        v8h a0 = *(const v8h*)(arow + k0);
        v8h a1 = *(const v8h*)(arow + k0 + 8);
        v16h av;
#pragma unroll
        for (int j = 0; j < 8; ++j) { av[j] = a0[j]; av[j + 8] = a1[j]; }
        v16h b0 = *(const v16h*)(brow + k0);
        v16h b1 = *(const v16h*)(brow + bstride + k0);
        v16h b2 = *(const v16h*)(brow + 2 * bstride + k0);
        v16h b3 = *(const v16h*)(brow + 3 * bstride + k0);
        acc0 = __builtin_amdgcn_wmma_f32_16x16x32_f16(false, av, false, b0, (short)0, acc0, false, false);
        acc1 = __builtin_amdgcn_wmma_f32_16x16x32_f16(false, av, false, b1, (short)0, acc1, false, false);
        acc2 = __builtin_amdgcn_wmma_f32_16x16x32_f16(false, av, false, b2, (short)0, acc2, false, false);
        acc3 = __builtin_amdgcn_wmma_f32_16x16x32_f16(false, av, false, b3, (short)0, acc3, false, false);
    }

    int mbase = mt * 16 + (kh << 3);
#pragma unroll
    for (int t = 0; t < 4; ++t) {
        v8f acc = (t == 0) ? acc0 : (t == 1) ? acc1 : (t == 2) ? acc2 : acc3;
        int ncol = ntg * 64 + t * 16 + l15;
        float bsv = bias ? bias[ncol] : 0.0f;
#pragma unroll
        for (int r = 0; r < 8; ++r) {
            int m = mbase + r;
            size_t idx = (size_t)m * ldc + ncol;
            float v = acc[r] + bsv;
            if (resid) v += (float)resid[idx];
            if (Cpre)  Cpre[idx] = (_Float16)v;
            if (do_relu && v < 0.0f) v = 0.0f;
            if (Cf) Cf[idx] = v;
            if (Ch) Ch[idx] = (_Float16)v;
        }
    }
}

// ----- staging: Bt[n*Kp+k] = (k<K) ? W[k*N+n] : 0   (fp32 -> f16, transpose) -----
__global__ __launch_bounds__(256)
void stage_wT(const float* __restrict__ W, _Float16* __restrict__ out,
              int K, int N, int Kp)
{
    int gid = blockIdx.x * 256 + threadIdx.x;
    if (gid >= N * Kp) return;
    int n = gid / Kp, k = gid - n * Kp;
    out[gid] = (k < K) ? (_Float16)W[(size_t)k * N + n] : (_Float16)0.0f;
}

// ----- staging: LSTM cat weights  Wcat[n*768 + k] = k<512 ? Wih[n][k] : Whh[n][k-512]
__global__ __launch_bounds__(256)
void stage_lstm(const float* __restrict__ Wih, const float* __restrict__ Whh,
                _Float16* __restrict__ out)
{
    int gid = blockIdx.x * 256 + threadIdx.x;   // 1024*768
    if (gid >= 1024 * 768) return;
    int n = gid / 768, k = gid - n * 768;
    float v = (k < 512) ? Wih[(size_t)n * 512 + k] : Whh[(size_t)n * 256 + (k - 512)];
    out[gid] = (_Float16)v;
}

// ----- staging: f_bonds fp32[NB1 x 147] -> f16[MB_PAD x 160], zero-padded -----
__global__ __launch_bounds__(256)
void stage_fbonds(const float* __restrict__ fb, _Float16* __restrict__ out)
{
    int gid = blockIdx.x * 256 + threadIdx.x;   // MB_PAD*KB_PAD
    if (gid >= MB_PAD * KB_PAD) return;
    int r = gid / KB_PAD, k = gid - r * KB_PAD;
    float v = (r < NB1 && k < BFD) ? fb[(size_t)r * BFD + k] : 0.0f;
    out[gid] = (_Float16)v;
}

// ----- a_message[a][c] = sum_j message[a2b[a][j]][c]  (one block per atom) -----
__global__ __launch_bounds__(256)
void gather_sum(const _Float16* __restrict__ msg, const int* __restrict__ a2b,
                _Float16* __restrict__ out)
{
    int a = blockIdx.x;            // 65536 blocks
    int c = threadIdx.x;           // 256
    float s = 0.0f;
#pragma unroll
    for (int j = 0; j < MAX_NB; ++j) {
        int b = a2b[a * MAX_NB + j];
        s += (float)msg[(size_t)b * HH + c];
    }
    out[(size_t)a * HH + c] = (_Float16)s;
}

// ----- bondin[b][c] = a_message[b2a[b]][c] - message[b2revb[b]][c] -----
__global__ __launch_bounds__(256)
void bond_input(const _Float16* __restrict__ amsg, const _Float16* __restrict__ msg,
                const int* __restrict__ b2a, const int* __restrict__ b2revb,
                _Float16* __restrict__ out)
{
    int b = blockIdx.x;            // MB_PAD blocks
    int c = threadIdx.x;
    float v = 0.0f;
    if (b < NB1) {
        v = (float)amsg[(size_t)b2a[b] * HH + c] - (float)msg[(size_t)b2revb[b] * HH + c];
    }
    out[(size_t)b * HH + c] = (_Float16)v;
}

__global__ void zero_row(_Float16* p) { p[threadIdx.x] = (_Float16)0.0f; }

// ----- concat[a] = [ f_atoms[a] (133) | nei[a] (256) | 0 pad (27) ] -----
__global__ __launch_bounds__(256)
void concat_build(const float* __restrict__ fa, const _Float16* __restrict__ nei,
                  _Float16* __restrict__ out)
{
    int gid = blockIdx.x * 256 + threadIdx.x;   // N_ATOMS*KO_PAD
    if (gid >= N_ATOMS * KO_PAD) return;
    int a = gid / KO_PAD, c = gid - a * KO_PAD;
    float v;
    if (c < AFD)            v = fa[(size_t)a * AFD + c];
    else if (c < AFD + HH)  v = (float)nei[(size_t)a * HH + (c - AFD)];
    else                    v = 0.0f;
    out[gid] = (_Float16)v;
}

// ----- LSTM gate nonlinearity; h written to qh cols [0,256) and [512,768) -----
__global__ __launch_bounds__(256)
void lstm_update(const float* __restrict__ gates, float* __restrict__ c,
                 _Float16* __restrict__ qh)
{
    int gid = blockIdx.x * 256 + threadIdx.x;   // Mrows*256 (grid sized by caller)
    int m = gid >> 8, d = gid & 255;
    const float* g = gates + (size_t)m * 1024;
    float ig = sigm(g[d]);
    float fg = sigm(g[256 + d]);
    float gg = tanhf(g[512 + d]);
    float og = sigm(g[768 + d]);
    float cn = fg * c[gid] + ig * gg;
    c[gid] = cn;
    _Float16 h = (_Float16)(og * tanhf(cn));
    qh[(size_t)m * 768 + d] = h;
    qh[(size_t)m * 768 + 512 + d] = h;
}

// ----- Set2Set attention: alpha=softmax(q . feat_n); readout -> qh cols [256,512) -----
__global__ __launch_bounds__(256)
void s2s_attention(const _Float16* __restrict__ feat, _Float16* __restrict__ qh, int N)
{
    __shared__ float part[256];
    __shared__ float dots[32];
    __shared__ float alpha[32];
    int m = blockIdx.x;
    int t = threadIdx.x;
    const _Float16* f = feat + (size_t)m * N * HH;
    const _Float16* q = qh + (size_t)m * 768;      // q = h (cols 0..255)

    float p = 0.0f;
    if (t < N * 8) {
        int n = t >> 3, ch = t & 7;
        const _Float16* fr = f + n * HH + ch * 32;
        const _Float16* qr = q + ch * 32;
#pragma unroll
        for (int d = 0; d < 32; ++d) p += (float)qr[d] * (float)fr[d];
    }
    part[t] = p;
    __syncthreads();
    if (t < N) {
        float s = 0.0f;
#pragma unroll
        for (int j = 0; j < 8; ++j) s += part[t * 8 + j];
        dots[t] = s;
    }
    __syncthreads();
    if (t == 0) {
        float mx = dots[0];
        for (int n = 1; n < N; ++n) mx = fmaxf(mx, dots[n]);
        float se = 0.0f;
        for (int n = 0; n < N; ++n) { float e = expf(dots[n] - mx); alpha[n] = e; se += e; }
        float inv = 1.0f / se;
        for (int n = 0; n < N; ++n) alpha[n] *= inv;
    }
    __syncthreads();
    float r = 0.0f;
    for (int n = 0; n < N; ++n) r += alpha[n] * (float)f[n * HH + t];
    qh[(size_t)m * 768 + 256 + t] = (_Float16)r;
}

// ----- NNAttention pair rows: row (b,s1,s2) = [mol[b,s1] | mol[b,s2]] -----
__global__ __launch_bounds__(256)
void pair_build(const _Float16* __restrict__ mol, _Float16* __restrict__ out)
{
    int gid = blockIdx.x * 256 + threadIdx.x;   // 8192*512
    int r = gid >> 9, col = gid & 511;
    int b = r >> 4, s1 = (r >> 2) & 3, s2 = r & 3;
    _Float16 v = (col < 256) ? mol[(size_t)(b * 4 + s1) * HH + col]
                             : mol[(size_t)(b * 4 + s2) * HH + (col - 256)];
    out[gid] = v;
}

// ----- diagonal replacement + reshape to [2048 x 1024] for NN1 -----
__global__ __launch_bounds__(256)
void combine_attn(const _Float16* __restrict__ pairO, const _Float16* __restrict__ selfO,
                  _Float16* __restrict__ out)
{
    int gid = blockIdx.x * 256 + threadIdx.x;   // 2048*1024
    int row = gid >> 10, col = gid & 1023;      // row = b*4+s1
    int s2 = col >> 8, d = col & 255, s1 = row & 3;
    out[gid] = (s1 == s2) ? selfO[(size_t)row * HH + d]
                          : pairO[((size_t)row * 4 + s2) * HH + d];
}

// ---------------------------------------------------------------------------
static inline void launch_gemm(hipStream_t s, const _Float16* A, int lda,
                               const _Float16* Bt, const float* bias,
                               const _Float16* resid, float* Cf, _Float16* Ch,
                               _Float16* Cpre, int ldc, int Mt, int Nt, int K, int relu)
{
    int waves = Mt * (Nt >> 2);        // 4 N-tiles per wave
    int blocks = (waves + 7) / 8;
    gemm_wmma<<<blocks, 256, 0, s>>>(A, lda, Bt, bias, resid, Cf, Ch, Cpre,
                                     ldc, Mt, Nt, K, relu);
}

extern "C" void kernel_launch(void* const* d_in, const int* in_sizes, int n_in,
                              void* d_out, int out_size, void* d_ws, size_t ws_size,
                              hipStream_t stream)
{
    const float* f_atoms      = (const float*)d_in[0];
    const float* f_bonds      = (const float*)d_in[1];
    const int*   a2b          = (const int*)  d_in[2];
    const int*   b2a          = (const int*)  d_in[3];
    const int*   b2revb       = (const int*)  d_in[4];
    const float* W_i          = (const float*)d_in[5];
    const float* W_h          = (const float*)d_in[6];
    const float* W_o          = (const float*)d_in[7];
    const float* b_o          = (const float*)d_in[8];
    const float* W_nn0        = (const float*)d_in[9];
    const float* b_nn0        = (const float*)d_in[10];
    const float* W_nn0s       = (const float*)d_in[11];
    const float* b_nn0s       = (const float*)d_in[12];
    const float* W_nn1        = (const float*)d_in[13];
    const float* b_nn1        = (const float*)d_in[14];
    const float* lstm_n_Wih   = (const float*)d_in[15];
    const float* lstm_n_Whh   = (const float*)d_in[16];
    const float* lstm_n_b     = (const float*)d_in[17];
    const float* node_cond_W  = (const float*)d_in[18];
    const float* node_cond_b  = (const float*)d_in[19];
    const float* lstm_g_Wih   = (const float*)d_in[20];
    const float* lstm_g_Whh   = (const float*)d_in[21];
    const float* lstm_g_b     = (const float*)d_in[22];
    const float* graph_cond_W = (const float*)d_in[23];
    const float* graph_cond_b = (const float*)d_in[24];
    float* out = (float*)d_out;

    // ---- carve workspace (~320 MB) ----
    char* base = (char*)d_ws;
    size_t off = 0;
    auto carve = [&](size_t bytes) -> void* {
        void* r = base + off;
        off += (bytes + 255) & ~(size_t)255;
        return r;
    };
    _Float16* fb16    = (_Float16*)carve((size_t)MB_PAD * KB_PAD * 2); // later reused as atomh16
    _Float16* inp16   = (_Float16*)carve((size_t)MB_PAD * HH * 2);
    _Float16* msg16   = (_Float16*)carve((size_t)MB_PAD * HH * 2);
    _Float16* tmpB16  = (_Float16*)carve((size_t)MB_PAD * HH * 2);     // later reused as concat16
    _Float16* amsg16  = (_Float16*)carve((size_t)N_ATOMS * HH * 2);
    _Float16* WiT     = (_Float16*)carve((size_t)HH * KB_PAD * 2);
    _Float16* WhT     = (_Float16*)carve((size_t)HH * HH * 2);
    _Float16* WoT     = (_Float16*)carve((size_t)HH * KO_PAD * 2);
    _Float16* Wnn0T   = (_Float16*)carve((size_t)HH * 512 * 2);
    _Float16* Wnn0sT  = (_Float16*)carve((size_t)HH * HH * 2);
    _Float16* Wnn1T   = (_Float16*)carve((size_t)HH * 1024 * 2);
    _Float16* ncondT  = (_Float16*)carve((size_t)HH * 512 * 2);
    _Float16* gcondT  = (_Float16*)carve((size_t)HH * 512 * 2);
    _Float16* WcatN   = (_Float16*)carve((size_t)1024 * 768 * 2);
    _Float16* WcatG   = (_Float16*)carve((size_t)1024 * 768 * 2);
    _Float16* qhN     = (_Float16*)carve((size_t)N_MOLS * 768 * 2);
    float*    gatesN  = (float*)   carve((size_t)N_MOLS * 1024 * 4);
    float*    cN      = (float*)   carve((size_t)N_MOLS * HH * 4);
    _Float16* mol16   = (_Float16*)carve((size_t)N_MOLS * HH * 2);
    _Float16* pairA16 = (_Float16*)carve((size_t)8192 * 512 * 2);
    _Float16* pairO16 = (_Float16*)carve((size_t)8192 * HH * 2);
    _Float16* selfO16 = (_Float16*)carve((size_t)N_MOLS * HH * 2);
    _Float16* nn1A16  = (_Float16*)carve((size_t)N_MOLS * 1024 * 2);
    _Float16* steps16 = (_Float16*)carve((size_t)N_MOLS * HH * 2);
    _Float16* qhG     = (_Float16*)carve((size_t)BB * 768 * 2);
    float*    gatesG  = (float*)   carve((size_t)BB * 1024 * 4);
    float*    cG      = (float*)   carve((size_t)BB * HH * 4);
    _Float16* atomh16  = fb16;     // alias: fb16 dead after first GEMM
    _Float16* concat16 = tmpB16;   // alias: tmpB16 dead after last message GEMM

    // ---- stage weights / inputs to f16 ----
    stage_fbonds<<<(MB_PAD * KB_PAD + 255) / 256, 256, 0, stream>>>(f_bonds, fb16);
    stage_wT<<<(HH * KB_PAD + 255) / 256, 256, 0, stream>>>(W_i, WiT, BFD, HH, KB_PAD);
    stage_wT<<<(HH * HH + 255) / 256, 256, 0, stream>>>(W_h, WhT, HH, HH, HH);
    stage_wT<<<(HH * KO_PAD + 255) / 256, 256, 0, stream>>>(W_o, WoT, AFD + HH, HH, KO_PAD);
    stage_wT<<<(HH * 512 + 255) / 256, 256, 0, stream>>>(W_nn0, Wnn0T, 512, HH, 512);
    stage_wT<<<(HH * HH + 255) / 256, 256, 0, stream>>>(W_nn0s, Wnn0sT, HH, HH, HH);
    stage_wT<<<(HH * 1024 + 255) / 256, 256, 0, stream>>>(W_nn1, Wnn1T, 1024, HH, 1024);
    stage_wT<<<(HH * 512 + 255) / 256, 256, 0, stream>>>(node_cond_W, ncondT, 512, HH, 512);
    stage_wT<<<(HH * 512 + 255) / 256, 256, 0, stream>>>(graph_cond_W, gcondT, 512, HH, 512);
    stage_lstm<<<(1024 * 768 + 255) / 256, 256, 0, stream>>>(lstm_n_Wih, lstm_n_Whh, WcatN);
    stage_lstm<<<(1024 * 768 + 255) / 256, 256, 0, stream>>>(lstm_g_Wih, lstm_g_Whh, WcatG);
    hipMemsetAsync(qhN, 0, (size_t)N_MOLS * 768 * 2, stream);
    hipMemsetAsync(cN, 0, (size_t)N_MOLS * HH * 4, stream);
    hipMemsetAsync(qhG, 0, (size_t)BB * 768 * 2, stream);
    hipMemsetAsync(cG, 0, (size_t)BB * HH * 4, stream);

    const int MtB = MB_PAD / 16;   // 8193 bond tiles
    const int MtA = N_ATOMS / 16;  // 4096 atom tiles

    // ---- inp = f_bonds @ W_i ; message = relu(inp) ----
    launch_gemm(stream, fb16, KB_PAD, WiT, nullptr, nullptr,
                nullptr, msg16, inp16, HH, MtB, 16, KB_PAD, 1);
    zero_row<<<1, 256, 0, stream>>>(msg16);

    // ---- depth-1 message passing iterations ----
    for (int it = 0; it < DEPTH - 1; ++it) {
        gather_sum<<<N_ATOMS, 256, 0, stream>>>(msg16, a2b, amsg16);
        bond_input<<<MB_PAD, 256, 0, stream>>>(amsg16, msg16, b2a, b2revb, tmpB16);
        launch_gemm(stream, tmpB16, HH, WhT, nullptr, inp16,
                    nullptr, msg16, nullptr, HH, MtB, 16, HH, 1);
        zero_row<<<1, 256, 0, stream>>>(msg16);
    }

    // ---- atom hidden: relu([f_atoms | nei] @ W_o + b_o) ----
    gather_sum<<<N_ATOMS, 256, 0, stream>>>(msg16, a2b, amsg16);
    concat_build<<<(N_ATOMS * KO_PAD + 255) / 256, 256, 0, stream>>>(f_atoms, amsg16, concat16);
    launch_gemm(stream, concat16, KO_PAD, WoT, b_o, nullptr,
                nullptr, atomh16, nullptr, HH, MtA, 16, KO_PAD, 1);

    // ---- node Set2Set (M=2048, N=32) ----
    for (int it = 0; it < N_ITERS; ++it) {
        launch_gemm(stream, qhN, 768, WcatN, lstm_n_b, nullptr,
                    gatesN, nullptr, nullptr, 1024, N_MOLS / 16, 64, 768, 0);
        lstm_update<<<N_MOLS * HH / 256, 256, 0, stream>>>(gatesN, cN, qhN);
        s2s_attention<<<N_MOLS, 256, 0, stream>>>(atomh16, qhN, APM);
    }
    launch_gemm(stream, qhN, 768, ncondT, node_cond_b, nullptr,
                nullptr, mol16, nullptr, HH, N_MOLS / 16, 16, 512, 0);

    // ---- NNAttention over steps ----
    pair_build<<<(8192 * 512) / 256, 256, 0, stream>>>(mol16, pairA16);
    launch_gemm(stream, pairA16, 512, Wnn0T, b_nn0, nullptr,
                nullptr, pairO16, nullptr, HH, 8192 / 16, 16, 512, 0);
    launch_gemm(stream, mol16, HH, Wnn0sT, b_nn0s, nullptr,
                nullptr, selfO16, nullptr, HH, N_MOLS / 16, 16, HH, 0);
    combine_attn<<<(N_MOLS * 1024) / 256, 256, 0, stream>>>(pairO16, selfO16, nn1A16);
    launch_gemm(stream, nn1A16, 1024, Wnn1T, b_nn1, nullptr,
                nullptr, steps16, nullptr, HH, N_MOLS / 16, 16, 1024, 0);

    // ---- graph Set2Set (M=512, N=4) ----
    for (int it = 0; it < N_ITERS; ++it) {
        launch_gemm(stream, qhG, 768, WcatG, lstm_g_b, nullptr,
                    gatesG, nullptr, nullptr, 1024, BB / 16, 64, 768, 0);
        lstm_update<<<BB * HH / 256, 256, 0, stream>>>(gatesG, cG, qhG);
        s2s_attention<<<BB, 256, 0, stream>>>(steps16, qhG, SS);
    }
    launch_gemm(stream, qhG, 768, gcondT, graph_cond_b, nullptr,
                out, nullptr, nullptr, HH, BB / 16, 16, 512, 0);
}
